// JiTBlock_90812788506929
// MI455X (gfx1250) — matrix operand order repaired and checked by reading, not compiled, not run on hardware
//
#include <hip/hip_runtime.h>
#include <hip/hip_bf16.h>

typedef __bf16 bf16_t;
typedef __attribute__((ext_vector_type(16))) __bf16 bf16x16;
typedef __attribute__((ext_vector_type(8)))  float   f32x8;
typedef int v4i_async __attribute__((vector_size(16)));
typedef __attribute__((address_space(1))) v4i_async* g_v4i_ptr;
typedef __attribute__((address_space(3))) v4i_async* l_v4i_ptr;

#define WMMA_BF16(a, b, c) \
  __builtin_amdgcn_wmma_f32_16x16x32_bf16(false, (a), false, (b), (short)0, (c), false, false)

#if defined(__has_builtin)
#  if __has_builtin(__builtin_amdgcn_global_load_async_to_lds_b128)
#    define HAVE_ASYNC_LDS 1
#  endif
#  if __has_builtin(__builtin_amdgcn_s_wait_asynccnt)
#    define HAVE_WAIT_ASYNC 1
#  endif
#endif

// 16B global -> LDS copy: async (ASYNCcnt-tracked, no VGPR staging) when the
// gfx1250 builtin exists, else plain load+ds_store.
static __device__ inline void async_copy16(const bf16_t* g, void* l) {
#if defined(HAVE_ASYNC_LDS)
  __builtin_amdgcn_global_load_async_to_lds_b128(
      (g_v4i_ptr)(g), (l_v4i_ptr)(l), 0, 0);
#else
  *(uint4*)l = *(const uint4*)g;
#endif
}

template <int N>
static __device__ inline void wait_async() {
#if defined(HAVE_ASYNC_LDS)
#  if defined(HAVE_WAIT_ASYNC)
  __builtin_amdgcn_s_wait_asynccnt(N);
#  else
  asm volatile("s_wait_asynccnt %0" ::"n"(N) : "memory");
#  endif
#endif
}

union ABFrag { uint4 u4[2]; bf16x16 v; };

// ---------------------------------------------------------------------------
// LDS fragment loaders (CDNA5 ISA 7.12.2 layouts; padded row strides, 16B
// aligned so each fragment is 2x ds_load_b128).
// ---------------------------------------------------------------------------
static __device__ inline bf16x16 ld_afrag_lds(const char* base, int strideB, int row0, int lane) {
  int half = lane >> 4, mr = lane & 15;
  const char* p = base + (size_t)(row0 + mr) * strideB + half * 16;
  ABFrag f;
  f.u4[0] = *(const uint4*)(p);
  f.u4[1] = *(const uint4*)(p + 32);
  return f.v;
}
static __device__ inline bf16x16 ld_bfrag_lds(const char* base, int strideB, int row0, int lane) {
  int half = lane >> 4, nr = lane & 15;
  const char* p = base + (size_t)(row0 + nr) * strideB + half * 16;
  ABFrag f;
  f.u4[0] = *(const uint4*)(p);
  f.u4[1] = *(const uint4*)(p + 32);
  return f.v;
}

// ---------------------------------------------------------------------------
__global__ __launch_bounds__(256) void cvt_pad_kernel(
    const float* __restrict__ src, bf16_t* __restrict__ dst,
    int rows, int cols, int prow, int pcol) {
  size_t i = (size_t)blockIdx.x * 256 + threadIdx.x;
  size_t tot = (size_t)prow * pcol;
  if (i >= tot) return;
  int r = (int)(i / pcol);
  int c = (int)(i % pcol);
  float v = (r < rows && c < cols) ? src[(size_t)r * cols + c] : 0.0f;
  dst[i] = (bf16_t)v;
}

// ---------------------------------------------------------------------------
__global__ __launch_bounds__(256) void adaln_kernel(
    const float* __restrict__ c, const float* __restrict__ ada_w,
    const float* __restrict__ ada_b, float* __restrict__ mods) {
  int o = blockIdx.x * 256 + threadIdx.x;
  if (o >= 2 * 9216) return;
  int b = o / 9216, col = o % 9216;
  const float* cr = c + (size_t)b * 1024;
  const float* wr = ada_w + (size_t)col * 1024;
  float acc = ada_b[col];
  for (int k = 0; k < 1024; ++k) {
    float cv = cr[k];
    acc += (cv / (1.0f + __expf(-cv))) * wr[k];
  }
  mods[o] = acc;
}

// ---------------------------------------------------------------------------
__global__ __launch_bounds__(256) void rmsnorm_mod_kernel(
    const float* __restrict__ src, const float* __restrict__ w,
    const float* __restrict__ mods, int sc_idx, int sh_idx,
    bf16_t* __restrict__ out, int mode) {
  const int row = blockIdx.x;
  const int tid = threadIdx.x;
  const int b = row >> 12;
  int srow = row;
  if (mode == 1) {
    int rem = row & 4095;
    int l = rem >> 4;
    int t = row & 15;
    srow = (b * 16 + t) * 256 + l;
  }
  const float* x = src + (size_t)srow * 1024;
  float4 xv = ((const float4*)x)[tid];
  float ss = xv.x * xv.x + xv.y * xv.y + xv.z * xv.z + xv.w * xv.w;
#pragma unroll
  for (int off = 16; off >= 1; off >>= 1) ss += __shfl_xor(ss, off, 32);
  __shared__ float red[8];
  if ((tid & 31) == 0) red[tid >> 5] = ss;
  __syncthreads();
  float tot = 0.0f;
#pragma unroll
  for (int i = 0; i < 8; ++i) tot += red[i];
  float r = rsqrtf(tot * (1.0f / 1024.0f) + 1e-6f);
  const float* sc = mods + (size_t)(b * 9 + sc_idx) * 1024;
  const float* sh = mods + (size_t)(b * 9 + sh_idx) * 1024;
  float vx[4] = {xv.x, xv.y, xv.z, xv.w};
#pragma unroll
  for (int j = 0; j < 4; ++j) {
    int n = tid * 4 + j;
    float val = vx[j] * r * w[n] * (1.0f + sc[n]) + sh[n];
    out[(size_t)row * 1024 + n] = (bf16_t)val;
  }
}

// ---------------------------------------------------------------------------
// bf16 GEMM, 128x128x32 tiles, 8 waves, async double-buffered LDS staging.
// mode 0: out fp32 + bias   mode 3: outb bf16 + bias
// mode 1: resout = resid + gate*(acc+bias)
// mode 2: like 1, with (b,l,t) -> (b,t,l) row remap
// ---------------------------------------------------------------------------
__global__ __launch_bounds__(256) void gemm_bf16_kernel(
    const bf16_t* __restrict__ A, const bf16_t* __restrict__ W,
    const float* __restrict__ bias, int biasN,
    float* __restrict__ out, bf16_t* __restrict__ outb, int ldc,
    const float* __restrict__ resid, float* __restrict__ resout,
    const float* __restrict__ mods, int gate_idx,
    int K, int mode) {
  __shared__ __align__(16) char sA[2][128 * 80];
  __shared__ __align__(16) char sB[2][128 * 80];
  const int tid = threadIdx.x;
  const int lane = tid & 31;
  const int wave = tid >> 5;
  const int wr = wave >> 2;  // 0..1  (64 rows each)
  const int wc = wave & 3;   // 0..3  (32 cols each)
  const int m0 = blockIdx.y * 128;
  const int n0 = blockIdx.x * 128;
  const int r0 = tid >> 2, cc0 = tid & 3;  // 16B chunk coords (128 rows x 4)

  auto issue = [&](int buf, int k0) {
#pragma unroll
    for (int i = 0; i < 2; ++i) {
      int r = r0 + i * 64;
      async_copy16(A + (size_t)(m0 + r) * K + k0 + cc0 * 8, &sA[buf][r * 80 + cc0 * 16]);
      async_copy16(W + (size_t)(n0 + r) * K + k0 + cc0 * 8, &sB[buf][r * 80 + cc0 * 16]);
    }
  };

  f32x8 acc[4][2] = {};
  const int nk = K >> 5;
  issue(0, 0);
  for (int kt = 0; kt < nk; ++kt) {
    const int buf = kt & 1;
    wait_async<0>();
    __syncthreads();
    if (kt + 1 < nk) issue(buf ^ 1, (kt + 1) << 5);

    bf16x16 af[4], bfr[2];
#pragma unroll
    for (int mt = 0; mt < 4; ++mt) af[mt] = ld_afrag_lds(sA[buf], 80, wr * 64 + mt * 16, lane);
#pragma unroll
    for (int nt = 0; nt < 2; ++nt) bfr[nt] = ld_bfrag_lds(sB[buf], 80, wc * 32 + nt * 16, lane);
#pragma unroll
    for (int mt = 0; mt < 4; ++mt)
#pragma unroll
      for (int nt = 0; nt < 2; ++nt)
        acc[mt][nt] = WMMA_BF16(af[mt], bfr[nt], acc[mt][nt]);
  }

  const int half = lane >> 4, ln = lane & 15;
#pragma unroll
  for (int mt = 0; mt < 4; ++mt) {
#pragma unroll
    for (int nt = 0; nt < 2; ++nt) {
      int n = n0 + wc * 32 + nt * 16 + ln;
      float bv = (n < biasN) ? bias[n] : 0.0f;
#pragma unroll
      for (int v = 0; v < 8; ++v) {
        int m = m0 + wr * 64 + mt * 16 + v + half * 8;
        float val = acc[mt][nt][v] + bv;
        if (mode == 0) {
          out[(size_t)m * ldc + n] = val;
        } else if (mode == 3) {
          outb[(size_t)m * ldc + n] = (bf16_t)val;
        } else {
          int b = m >> 12;
          int row = m;
          if (mode == 2) {
            int rem = m & 4095;
            int l = rem >> 4;
            int t = m & 15;
            row = (b * 16 + t) * 256 + l;
          }
          float g = mods[(size_t)(b * 9 + gate_idx) * 1024 + n];
          resout[(size_t)row * 1024 + n] = resid[(size_t)row * 1024 + n] + g * val;
        }
      }
    }
  }
}

// ---------------------------------------------------------------------------
// Per-head rmsnorm + RoPE on q,k; v repacked (transposed) for B-fragments.
// ---------------------------------------------------------------------------
__global__ __launch_bounds__(256) void qk_prep_kernel(
    const float* __restrict__ qkv, const float* __restrict__ cs,
    const float* __restrict__ sn, const float* __restrict__ qn_w,
    const float* __restrict__ kn_w, bf16_t* __restrict__ qb,
    bf16_t* __restrict__ kb, bf16_t* __restrict__ vtb, int temporal) {
  int gid = blockIdx.x * 256 + threadIdx.x;
  int lane = threadIdx.x & 31;
  int wid = gid >> 5;
  int h = wid & 15;
  int m = wid >> 4;
  const float* base = qkv + (size_t)m * 3072 + h * 64;
  float q0 = base[lane], q1 = base[lane + 32];
  float k0 = base[1024 + lane], k1 = base[1024 + lane + 32];
  float v0 = base[2048 + lane], v1 = base[2048 + lane + 32];
  float ssq = q0 * q0 + q1 * q1;
  float ssk = k0 * k0 + k1 * k1;
#pragma unroll
  for (int off = 16; off >= 1; off >>= 1) {
    ssq += __shfl_xor(ssq, off, 32);
    ssk += __shfl_xor(ssk, off, 32);
  }
  float rq = rsqrtf(ssq * (1.0f / 64.0f) + 1e-6f);
  float rk = rsqrtf(ssk * (1.0f / 64.0f) + 1e-6f);
  int seq, pos;
  if (temporal) { seq = m >> 4; pos = m & 15; }
  else          { seq = m >> 8; pos = m & 255; }
  float c0 = cs[pos * 64 + lane], c1 = cs[pos * 64 + lane + 32];
  float s0 = sn[pos * 64 + lane], s1 = sn[pos * 64 + lane + 32];
  float qa = q0 * rq * qn_w[lane], qc = q1 * rq * qn_w[lane + 32];
  float ka = k0 * rk * kn_w[lane], kc = k1 * rk * kn_w[lane + 32];
  float qo0 = qa * c0 - qc * s0;
  float qo1 = qc * c1 + qa * s1;
  float ko0 = ka * c0 - kc * s0;
  float ko1 = kc * c1 + ka * s1;
  int sh_i = seq * 16 + h;
  int len = temporal ? 16 : 256;
  size_t qoff = ((size_t)sh_i * len + pos) * 64;
  qb[qoff + lane] = (bf16_t)qo0;
  qb[qoff + lane + 32] = (bf16_t)qo1;
  kb[qoff + lane] = (bf16_t)ko0;
  kb[qoff + lane + 32] = (bf16_t)ko1;
  if (temporal) {
    size_t voff = (size_t)sh_i * 64 * 32;
    vtb[voff + (size_t)lane * 32 + pos] = (bf16_t)v0;
    vtb[voff + (size_t)(lane + 32) * 32 + pos] = (bf16_t)v1;
    vtb[voff + (size_t)lane * 32 + pos + 16] = (bf16_t)0.0f;
    vtb[voff + (size_t)(lane + 32) * 32 + pos + 16] = (bf16_t)0.0f;
  } else {
    size_t voff = (size_t)sh_i * 64 * 256;
    vtb[voff + (size_t)lane * 256 + pos] = (bf16_t)v0;
    vtb[voff + (size_t)(lane + 32) * 256 + pos] = (bf16_t)v1;
  }
}

// ---------------------------------------------------------------------------
// Spatial attention (L=256, HD=64): flash-style, async-staged K/V^T blocks.
// ---------------------------------------------------------------------------
__global__ __launch_bounds__(128) void attn_spatial_kernel(
    const bf16_t* __restrict__ qb, const bf16_t* __restrict__ kb,
    const bf16_t* __restrict__ vtb, bf16_t* __restrict__ aout) {
  __shared__ __align__(16) char sK[64 * 144];
  __shared__ __align__(16) char sV[64 * 144];
  __shared__ __align__(16) char sP[4 * 16 * 144];
  const int bh = blockIdx.x >> 2;
  const int qc = blockIdx.x & 3;
  const int tid = threadIdx.x;
  const int wave = tid >> 5;
  const int lane = tid & 31;
  const int half = lane >> 4, ln = lane & 15;
  const int qr0 = qc * 64 + wave * 16;
  const bf16_t* Q = qb + (size_t)bh * 256 * 64;
  const bf16_t* Kp = kb + (size_t)bh * 256 * 64;
  const bf16_t* Vt = vtb + (size_t)bh * 64 * 256;

  bf16x16 aq[2];
#pragma unroll
  for (int f = 0; f < 2; ++f) {
    ABFrag fr;
    const bf16_t* p = Q + (size_t)(qr0 + ln) * 64 + f * 32 + half * 8;
    fr.u4[0] = *(const uint4*)(p);
    fr.u4[1] = *(const uint4*)(p + 16);
    aq[f] = fr.v;
  }

  f32x8 o[4] = {};
  f32x8 rmax, rsum;
#pragma unroll
  for (int v = 0; v < 8; ++v) { rmax[v] = -3.0e38f; rsum[v] = 0.0f; }
  char* pw = sP + wave * 16 * 144;

  for (int kblk = 0; kblk < 4; ++kblk) {
    __syncthreads();
#pragma unroll
    for (int i = 0; i < 4; ++i) {
      int c = tid + i * 128;
      int r = c >> 3, cc = c & 7;
      async_copy16(Kp + (size_t)(kblk * 64 + r) * 64 + cc * 8, sK + r * 144 + cc * 16);
      async_copy16(Vt + (size_t)r * 256 + kblk * 64 + cc * 8, sV + r * 144 + cc * 16);
    }
    wait_async<0>();
    __syncthreads();

    // S = Q K^T * 0.125
    f32x8 s[4];
#pragma unroll
    for (int nt = 0; nt < 4; ++nt) {
      f32x8 a_ = {};
#pragma unroll
      for (int f = 0; f < 2; ++f) {
        ABFrag fr;
        const char* p = sK + (size_t)(nt * 16 + ln) * 144 + f * 64 + half * 16;
        fr.u4[0] = *(const uint4*)(p);
        fr.u4[1] = *(const uint4*)(p + 32);
        a_ = WMMA_BF16(aq[f], fr.v, a_);
      }
#pragma unroll
      for (int v = 0; v < 8; ++v) s[nt][v] = a_[v] * 0.125f;
    }

    // online softmax statistics
    f32x8 bm;
#pragma unroll
    for (int v = 0; v < 8; ++v)
      bm[v] = fmaxf(fmaxf(s[0][v], s[1][v]), fmaxf(s[2][v], s[3][v]));
#pragma unroll
    for (int off = 1; off < 16; off <<= 1)
#pragma unroll
      for (int v = 0; v < 8; ++v) bm[v] = fmaxf(bm[v], __shfl_xor(bm[v], off, 32));
    f32x8 scalef, bs;
#pragma unroll
    for (int v = 0; v < 8; ++v) {
      float nm = fmaxf(rmax[v], bm[v]);
      scalef[v] = __expf(rmax[v] - nm);
      rmax[v] = nm;
      bs[v] = 0.0f;
    }
#pragma unroll
    for (int nt = 0; nt < 4; ++nt)
#pragma unroll
      for (int v = 0; v < 8; ++v) {
        s[nt][v] = __expf(s[nt][v] - rmax[v]);
        bs[v] += s[nt][v];
      }
#pragma unroll
    for (int off = 1; off < 16; off <<= 1)
#pragma unroll
      for (int v = 0; v < 8; ++v) bs[v] += __shfl_xor(bs[v], off, 32);
#pragma unroll
    for (int v = 0; v < 8; ++v) rsum[v] = rsum[v] * scalef[v] + bs[v];
#pragma unroll
    for (int dt = 0; dt < 4; ++dt)
#pragma unroll
      for (int v = 0; v < 8; ++v) o[dt][v] *= scalef[v];

    // stage P (C-layout -> A-layout via LDS)
#pragma unroll
    for (int nt = 0; nt < 4; ++nt)
#pragma unroll
      for (int v = 0; v < 8; ++v)
        *(bf16_t*)(pw + (size_t)(v + half * 8) * 144 + (nt * 16 + ln) * 2) =
            (bf16_t)s[nt][v];
    __syncthreads();

    // O += P V
#pragma unroll
    for (int dt = 0; dt < 4; ++dt) {
#pragma unroll
      for (int f = 0; f < 2; ++f) {
        ABFrag pa, vb_;
        const char* pp = pw + (size_t)ln * 144 + f * 64 + half * 16;
        pa.u4[0] = *(const uint4*)(pp);
        pa.u4[1] = *(const uint4*)(pp + 32);
        const char* vp = sV + (size_t)(dt * 16 + ln) * 144 + f * 64 + half * 16;
        vb_.u4[0] = *(const uint4*)(vp);
        vb_.u4[1] = *(const uint4*)(vp + 32);
        o[dt] = WMMA_BF16(pa.v, vb_.v, o[dt]);
      }
    }
  }

  const int bt = bh >> 4, h = bh & 15;
#pragma unroll
  for (int dt = 0; dt < 4; ++dt)
#pragma unroll
    for (int v = 0; v < 8; ++v) {
      int qrow = qr0 + v + half * 8;
      float val = o[dt][v] / rsum[v];
      aout[((size_t)(bt * 256 + qrow)) * 1024 + h * 64 + dt * 16 + ln] = (bf16_t)val;
    }
}

// ---------------------------------------------------------------------------
// Temporal attention (T=16, HD=64): one wave per (b,l,head).
// ---------------------------------------------------------------------------
__global__ __launch_bounds__(128) void attn_temporal_kernel(
    const bf16_t* __restrict__ qb, const bf16_t* __restrict__ kb,
    const bf16_t* __restrict__ vtb, bf16_t* __restrict__ aout) {
  __shared__ __align__(16) char sP[4 * 16 * 80];
  const int tid = threadIdx.x;
  const int wave = tid >> 5;
  const int lane = tid & 31;
  const int half = lane >> 4, ln = lane & 15;
  const int sh_i = blockIdx.x * 4 + wave;
  const bf16_t* Q = qb + (size_t)sh_i * 16 * 64;
  const bf16_t* Kp = kb + (size_t)sh_i * 16 * 64;
  const bf16_t* Vt = vtb + (size_t)sh_i * 64 * 32;

  f32x8 s = {};
#pragma unroll
  for (int f = 0; f < 2; ++f) {
    ABFrag qa, ka;
    const bf16_t* p = Q + (size_t)ln * 64 + f * 32 + half * 8;
    qa.u4[0] = *(const uint4*)(p);
    qa.u4[1] = *(const uint4*)(p + 16);
    const bf16_t* kp2 = Kp + (size_t)ln * 64 + f * 32 + half * 8;
    ka.u4[0] = *(const uint4*)(kp2);
    ka.u4[1] = *(const uint4*)(kp2 + 16);
    s = WMMA_BF16(qa.v, ka.v, s);
  }
#pragma unroll
  for (int v = 0; v < 8; ++v) s[v] *= 0.125f;
  f32x8 bm = s;
#pragma unroll
  for (int off = 1; off < 16; off <<= 1)
#pragma unroll
    for (int v = 0; v < 8; ++v) bm[v] = fmaxf(bm[v], __shfl_xor(bm[v], off, 32));
#pragma unroll
  for (int v = 0; v < 8; ++v) s[v] = __expf(s[v] - bm[v]);
  f32x8 bs = s;
#pragma unroll
  for (int off = 1; off < 16; off <<= 1)
#pragma unroll
    for (int v = 0; v < 8; ++v) bs[v] += __shfl_xor(bs[v], off, 32);
#pragma unroll
  for (int v = 0; v < 8; ++v) s[v] /= bs[v];

  char* pw = sP + wave * 16 * 80;
#pragma unroll
  for (int v = 0; v < 8; ++v) {
    char* rp = pw + (size_t)(v + half * 8) * 80;
    *(bf16_t*)(rp + ln * 2) = (bf16_t)s[v];
    *(bf16_t*)(rp + (ln + 16) * 2) = (bf16_t)0.0f;
  }
  asm volatile("s_wait_dscnt 0x0" ::: "memory");

  f32x8 o[4];
#pragma unroll
  for (int dt = 0; dt < 4; ++dt) {
    ABFrag pa, vb_;
    const char* pp = pw + (size_t)ln * 80 + half * 16;
    pa.u4[0] = *(const uint4*)(pp);
    pa.u4[1] = *(const uint4*)(pp + 32);
    const bf16_t* vp = Vt + (size_t)(dt * 16 + ln) * 32 + half * 8;
    vb_.u4[0] = *(const uint4*)(vp);
    vb_.u4[1] = *(const uint4*)(vp + 16);
    f32x8 z = {};
    o[dt] = WMMA_BF16(pa.v, vb_.v, z);
  }

  const int sidx = sh_i >> 4, h = sh_i & 15;
#pragma unroll
  for (int dt = 0; dt < 4; ++dt)
#pragma unroll
    for (int v = 0; v < 8; ++v) {
      int t = v + half * 8;
      aout[((size_t)(sidx * 16 + t)) * 1024 + h * 64 + dt * 16 + ln] =
          (bf16_t)o[dt][v];
    }
}

// ---------------------------------------------------------------------------
// SwiGLU on bf16 h12: hsw[m][j] = silu(h1)*h2, padded to 2752 cols.
// ---------------------------------------------------------------------------
__global__ __launch_bounds__(256) void swiglu_kernel(
    const bf16_t* __restrict__ h12b, bf16_t* __restrict__ hsw) {
  size_t i = (size_t)blockIdx.x * 256 + threadIdx.x;
  if (i >= (size_t)8192 * 2752) return;
  int m = (int)(i / 2752);
  int j = (int)(i % 2752);
  float r = 0.0f;
  if (j < 2730) {
    float a = (float)h12b[(size_t)m * 5504 + j];
    float b = (float)h12b[(size_t)m * 5504 + 2730 + j];
    r = (a / (1.0f + __expf(-a))) * b;
  }
  hsw[i] = (bf16_t)r;
}

// ---------------------------------------------------------------------------
extern "C" void kernel_launch(void* const* d_in, const int* in_sizes, int n_in,
                              void* d_out, int out_size, void* d_ws, size_t ws_size,
                              hipStream_t stream) {
  const float* x        = (const float*)d_in[0];
  const float* c        = (const float*)d_in[1];
  const float* cos_s    = (const float*)d_in[2];
  const float* sin_s    = (const float*)d_in[3];
  const float* cos_t    = (const float*)d_in[4];
  const float* sin_t    = (const float*)d_in[5];
  const float* norm1_w  = (const float*)d_in[6];
  const float* norm2_w  = (const float*)d_in[7];
  const float* norm3_w  = (const float*)d_in[8];
  const float* qkv_s_w  = (const float*)d_in[9];
  const float* qkv_s_b  = (const float*)d_in[10];
  const float* qn_s_w   = (const float*)d_in[11];
  const float* kn_s_w   = (const float*)d_in[12];
  const float* proj_s_w = (const float*)d_in[13];
  const float* proj_s_b = (const float*)d_in[14];
  const float* qkv_t_w  = (const float*)d_in[15];
  const float* qkv_t_b  = (const float*)d_in[16];
  const float* qn_t_w   = (const float*)d_in[17];
  const float* kn_t_w   = (const float*)d_in[18];
  const float* proj_t_w = (const float*)d_in[19];
  const float* proj_t_b = (const float*)d_in[20];
  const float* w12_w    = (const float*)d_in[21];
  const float* w12_b    = (const float*)d_in[22];
  const float* w3_w     = (const float*)d_in[23];
  const float* w3_b     = (const float*)d_in[24];
  const float* ada_w    = (const float*)d_in[25];
  const float* ada_b    = (const float*)d_in[26];
  float* out = (float*)d_out;
  (void)in_sizes; (void)n_in; (void)out_size; (void)ws_size;

  char* ws = (char*)d_ws;
  size_t off = 0;
  auto alloc = [&](size_t bytes) -> char* {
    char* p = ws + off;
    off += (bytes + 255) & ~(size_t)255;
    return p;
  };
  bf16_t* wb_qkv_s  = (bf16_t*)alloc((size_t)3072 * 1024 * 2);
  bf16_t* wb_qkv_t  = (bf16_t*)alloc((size_t)3072 * 1024 * 2);
  bf16_t* wb_proj_s = (bf16_t*)alloc((size_t)1024 * 1024 * 2);
  bf16_t* wb_proj_t = (bf16_t*)alloc((size_t)1024 * 1024 * 2);
  bf16_t* wb_w12    = (bf16_t*)alloc((size_t)5504 * 1024 * 2);
  bf16_t* wb_w3     = (bf16_t*)alloc((size_t)1024 * 2752 * 2);
  float*  mods      = (float*)alloc((size_t)2 * 9 * 1024 * 4);
  bf16_t* anbuf     = (bf16_t*)alloc((size_t)8192 * 1024 * 2);
  float*  qkvbuf    = (float*)alloc((size_t)8192 * 3072 * 4);
  bf16_t* qbuf      = (bf16_t*)alloc((size_t)8192 * 1024 * 2);
  bf16_t* kbuf      = (bf16_t*)alloc((size_t)8192 * 1024 * 2);
  bf16_t* vtbuf     = (bf16_t*)alloc((size_t)8192 * 64 * 32 * 2);
  bf16_t* aout      = (bf16_t*)alloc((size_t)8192 * 1024 * 2);
  float*  xbuf      = (float*)alloc((size_t)8192 * 1024 * 4);
  bf16_t* h12b      = (bf16_t*)alloc((size_t)8192 * 5504 * 2);
  bf16_t* hsw       = (bf16_t*)alloc((size_t)8192 * 2752 * 2);

  auto cvt = [&](const float* s, bf16_t* dq, int rows, int cols, int prow, int pcol) {
    size_t tot = (size_t)prow * pcol;
    cvt_pad_kernel<<<(unsigned)((tot + 255) / 256), 256, 0, stream>>>(s, dq, rows, cols, prow, pcol);
  };
  cvt(qkv_s_w,  wb_qkv_s,  3072, 1024, 3072, 1024);
  cvt(qkv_t_w,  wb_qkv_t,  3072, 1024, 3072, 1024);
  cvt(proj_s_w, wb_proj_s, 1024, 1024, 1024, 1024);
  cvt(proj_t_w, wb_proj_t, 1024, 1024, 1024, 1024);
  cvt(w12_w,    wb_w12,    5460, 1024, 5504, 1024);
  cvt(w3_w,     wb_w3,     1024, 2730, 1024, 2752);

  adaln_kernel<<<(2 * 9216 + 255) / 256, 256, 0, stream>>>(c, ada_w, ada_b, mods);

  // ---- spatial branch ----
  rmsnorm_mod_kernel<<<8192, 256, 0, stream>>>(x, norm1_w, mods, 1, 0, anbuf, 0);
  gemm_bf16_kernel<<<dim3(24, 64), 256, 0, stream>>>(
      anbuf, wb_qkv_s, qkv_s_b, 3072, qkvbuf, nullptr, 3072,
      nullptr, nullptr, nullptr, 0, 1024, 0);
  qk_prep_kernel<<<16384, 256, 0, stream>>>(
      qkvbuf, cos_s, sin_s, qn_s_w, kn_s_w, qbuf, kbuf, vtbuf, 0);
  attn_spatial_kernel<<<2048, 128, 0, stream>>>(qbuf, kbuf, vtbuf, aout);
  gemm_bf16_kernel<<<dim3(8, 64), 256, 0, stream>>>(
      aout, wb_proj_s, proj_s_b, 1024, nullptr, nullptr, 0,
      x, xbuf, mods, 2, 1024, 1);

  // ---- temporal branch ----
  rmsnorm_mod_kernel<<<8192, 256, 0, stream>>>(xbuf, norm2_w, mods, 4, 3, anbuf, 1);
  gemm_bf16_kernel<<<dim3(24, 64), 256, 0, stream>>>(
      anbuf, wb_qkv_t, qkv_t_b, 3072, qkvbuf, nullptr, 3072,
      nullptr, nullptr, nullptr, 0, 1024, 0);
  qk_prep_kernel<<<16384, 256, 0, stream>>>(
      qkvbuf, cos_t, sin_t, qn_t_w, kn_t_w, qbuf, kbuf, vtbuf, 1);
  attn_temporal_kernel<<<2048, 128, 0, stream>>>(qbuf, kbuf, vtbuf, aout);
  gemm_bf16_kernel<<<dim3(8, 64), 256, 0, stream>>>(
      aout, wb_proj_t, proj_t_b, 1024, nullptr, nullptr, 0,
      xbuf, xbuf, mods, 5, 1024, 2);

  // ---- MLP ----
  rmsnorm_mod_kernel<<<8192, 256, 0, stream>>>(xbuf, norm3_w, mods, 7, 6, anbuf, 2);
  gemm_bf16_kernel<<<dim3(43, 64), 256, 0, stream>>>(
      anbuf, wb_w12, w12_b, 5460, nullptr, h12b, 5504,
      nullptr, nullptr, nullptr, 0, 1024, 3);
  swiglu_kernel<<<(unsigned)(((size_t)8192 * 2752 + 255) / 256), 256, 0, stream>>>(h12b, hsw);
  gemm_bf16_kernel<<<dim3(8, 64), 256, 0, stream>>>(
      hsw, wb_w3, w3_b, 1024, nullptr, nullptr, 0,
      xbuf, out, mods, 8, 2752, 1);
}
